// Attention_56538949485240
// MI455X (gfx1250) — matrix operand order repaired and checked by reading, not compiled, not run on hardware
//
#include <hip/hip_runtime.h>
#include <hip/hip_bf16.h>
#include <cstdint>

typedef __attribute__((ext_vector_type(16))) __bf16 v16bf;
typedef __attribute__((ext_vector_type(8)))  float  v8f;
typedef __attribute__((ext_vector_type(4)))  __bf16 v4bf;

#define DIM      2048
#define HEAD_DIM 128
#define NUM_HEAD 16
#define BSZ      2
#define SEQ      2048
#define MROWS    (BSZ * SEQ)   // 4096

// ---------------------------------------------------------------------------
// CDNA5 async global->LDS copy (ASYNCcnt-tracked, bypasses VGPRs).
// Each lane copies 16B from its global address to its LDS offset.
// ---------------------------------------------------------------------------
static __device__ __forceinline__ void async_load_b128(const void* g, unsigned ldsOff) {
    asm volatile("global_load_async_to_lds_b128 %0, %1, off"
                 :: "v"(ldsOff), "v"(g) : "memory");
}
static __device__ __forceinline__ void wait_async0() {
    asm volatile("s_wait_asynccnt 0" ::: "memory");
}
static __device__ __forceinline__ unsigned lds_off(const void* p) {
    return (unsigned)(uintptr_t)p;   // low 32 bits of flat shared ptr = LDS offset
}

// ---------------------------------------------------------------------------
// WMMA fragment loaders from LDS (wave32, 16x16x32 bf16 shapes).
// A-matrix 16x32 (ISA 7.12.2): lanes 0-15 -> rows 0-15 holding K 0..7 (v0..3)
// and K 16..23 (v4..7); lanes 16-31 -> same rows, K 8..15 and 24..31.
// ---------------------------------------------------------------------------
static __device__ __forceinline__ v16bf frag_a(const __bf16* base, int stride, int kOff) {
    int lane = threadIdx.x & 31;
    const __bf16* p = base + (lane & 15) * stride + kOff + ((lane >> 4) << 3);
    union { v16bf v; uint4 u[2]; } r;
    r.u[0] = *(const uint4*)(p);
    r.u[1] = *(const uint4*)(p + 16);
    return r.v;
}

// B-matrix 32x16 stored as Bt[N][K]: lane n (0-15) holds column N=n with
// K 0..7 and 16..23; lanes 16-31 hold K 8..15 and 24..31 (same as A).
static __device__ __forceinline__ v16bf frag_b(const __bf16* base, int stride, int kOff) {
    int lane = threadIdx.x & 31;
    const __bf16* p = base + (lane & 15) * stride + kOff + ((lane >> 4) << 3);
    union { v16bf v; uint4 u[2]; } r;
    r.u[0] = *(const uint4*)(p);
    r.u[1] = *(const uint4*)(p + 16);
    return r.v;
}

static __device__ __forceinline__ v8f wmma_bf16(v16bf a, v16bf b, v8f c) {
    return __builtin_amdgcn_wmma_f32_16x16x32_bf16(false, a, false, b, (short)0, c, false, false);
}

// ---------------------------------------------------------------------------
// fp32 -> bf16 straight convert (vector of 4 per thread)
// ---------------------------------------------------------------------------
__global__ __launch_bounds__(256) void convert_f32_bf16(const float* __restrict__ in,
                                                        __bf16* __restrict__ out, int n) {
    int i = (blockIdx.x * 256 + threadIdx.x) * 4;
    if (i >= n) return;
    float4 v = *(const float4*)(in + i);
    v4bf o;
    o.x = (__bf16)v.x; o.y = (__bf16)v.y; o.z = (__bf16)v.z; o.w = (__bf16)v.w;
    *(v4bf*)(out + i) = o;
}

// ---------------------------------------------------------------------------
// Tiled 64x64 transpose + fp32->bf16 convert.
// Used for weights ([K,N]fp32 -> [N,K]bf16) and for V ([S,D] slice -> [D,S]).
// ---------------------------------------------------------------------------
__global__ __launch_bounds__(256) void transpose_conv(const float* __restrict__ src, size_t sStride,
                                                      __bf16* __restrict__ dst, size_t dStride,
                                                      int hCount, size_t srcBStride,
                                                      size_t srcHStride, size_t dstZStride) {
    __shared__ float tile[64][65];
    int z = blockIdx.z;
    int b = z / hCount, h = z % hCount;
    src += (size_t)b * srcBStride + (size_t)h * srcHStride;
    dst += (size_t)z * dstZStride;
    int r0 = blockIdx.x * 64, c0 = blockIdx.y * 64;
    int t = threadIdx.x;
#pragma unroll
    for (int i = 0; i < 16; ++i) {
        int lin = t + 256 * i;
        int sr = lin >> 6, dc = lin & 63;
        tile[sr][dc] = src[(size_t)(r0 + sr) * sStride + (c0 + dc)];
    }
    __syncthreads();
#pragma unroll
    for (int i = 0; i < 16; ++i) {
        int lin = t + 256 * i;
        int dr = lin >> 6, sc = lin & 63;
        dst[(size_t)(c0 + dr) * dStride + (r0 + sc)] = (__bf16)tile[sc][dr];
    }
}

// ---------------------------------------------------------------------------
// GEMM: C[M,N] = A[M,K](bf16) * Bt[N,K](bf16)^T + bias[N], fp32 out.
// WG tile 128x256, K-step 32, double-buffered LDS filled by async-to-LDS.
// 8 waves: 2 (M) x 4 (N); each wave: 64x64 = 4x4 wmma tiles (16 wmma/stage).
// ---------------------------------------------------------------------------
__global__ __launch_bounds__(256) void gemm_bf16(const __bf16* __restrict__ A,
                                                 const __bf16* __restrict__ Bt,
                                                 const float* __restrict__ bias,
                                                 float* __restrict__ C,
                                                 int M, int N, int K) {
    __shared__ __bf16 lA[2][128 * 32];   // 8 KB per stage
    __shared__ __bf16 lB[2][256 * 32];   // 16 KB per stage
    const int t = threadIdx.x;
    const int lane = t & 31;
    const int wave = t >> 5;
    const int wm = wave & 1;   // 0..1  (64 M rows each)
    const int wn = wave >> 1;  // 0..3  (64 N cols each)
    const int m0 = blockIdx.x * 128;
    const int n0 = blockIdx.y * 256;

    // Per-thread async chunk pointers (16B chunks), advanced 64B per stage.
    const __bf16* gA[2];
    const __bf16* gB[4];
    unsigned laOff[2], lbOff[4];
#pragma unroll
    for (int j = 0; j < 2; ++j) {                 // lA: 512 chunks
        int lin = t + j * 256;
        int row = lin >> 2, c8 = (lin & 3) * 8;
        gA[j] = A + (size_t)(m0 + row) * K + c8;
        laOff[j] = lds_off(&lA[0][lin * 8]);
    }
#pragma unroll
    for (int j = 0; j < 4; ++j) {                 // lB: 1024 chunks
        int lin = t + j * 256;
        int row = lin >> 2, c8 = (lin & 3) * 8;
        gB[j] = Bt + (size_t)(n0 + row) * K + c8;
        lbOff[j] = lds_off(&lB[0][lin * 8]);
    }

    v8f acc[4][4];
#pragma unroll
    for (int i = 0; i < 4; ++i)
#pragma unroll
        for (int j = 0; j < 4; ++j) acc[i][j] = {};

    // issue stage into buffer p, then advance pointers by K-step (32 halfs)
    auto issue = [&](int p) {
#pragma unroll
        for (int j = 0; j < 2; ++j) {
            async_load_b128(gA[j], laOff[j] + p * (128 * 32 * 2));
            gA[j] += 32;
        }
#pragma unroll
        for (int j = 0; j < 4; ++j) {
            async_load_b128(gB[j], lbOff[j] + p * (256 * 32 * 2));
            gB[j] += 32;
        }
    };

    issue(0);
    const int stages = K / 32;
    for (int s = 0; s < stages; ++s) {
        const int p = s & 1;
        wait_async0();
        __syncthreads();                 // stage s resident everywhere
        if (s + 1 < stages) issue(p ^ 1);

        v16bf af[4];
#pragma unroll
        for (int tm = 0; tm < 4; ++tm)
            af[tm] = frag_a(&lA[p][(wm * 64 + tm * 16) * 32], 32, 0);
#pragma unroll
        for (int tn = 0; tn < 4; ++tn) {
            v16bf bfr = frag_b(&lB[p][(wn * 64 + tn * 16) * 32], 32, 0);
#pragma unroll
            for (int tm = 0; tm < 4; ++tm)
                acc[tm][tn] = wmma_bf16(af[tm], bfr, acc[tm][tn]);
        }
    }

    const int colLane = lane & 15;
    const int rowSel = (lane >> 4) << 3;
#pragma unroll
    for (int tn = 0; tn < 4; ++tn) {
        int col = n0 + wn * 64 + tn * 16 + colLane;
        float b = bias[col];
#pragma unroll
        for (int tm = 0; tm < 4; ++tm) {
            int mBase = m0 + wm * 64 + tm * 16 + rowSel;
#pragma unroll
            for (int r = 0; r < 8; ++r)
                C[(size_t)(mBase + r) * N + col] = acc[tm][tn][r] + b;
        }
    }
}

// ---------------------------------------------------------------------------
// RoPE + convert + relayout: fp32 [B*S, H*D] -> bf16 [B,H,S,D]
// ---------------------------------------------------------------------------
__global__ __launch_bounds__(256) void rope_bf16(const float* __restrict__ X,
                                                 const float* __restrict__ cosT,
                                                 const float* __restrict__ sinT,
                                                 __bf16* __restrict__ Out) {
    int idx = blockIdx.x * 256 + threadIdx.x;  // B*S*H*D threads
    int d = idx & 127;
    int rest = idx >> 7;
    int h = rest & 15;
    int bs = rest >> 4;            // b*SEQ + s
    int s = bs & (SEQ - 1);
    int b = bs >> 11;
    const float* row = X + (size_t)bs * DIM + h * HEAD_DIM;
    float v = row[d];
    float partner = (d < 64) ? -row[d + 64] : row[d - 64];
    float out = v * cosT[s * HEAD_DIM + d] + partner * sinT[s * HEAD_DIM + d];
    Out[(((size_t)(b * NUM_HEAD + h)) * SEQ + s) * HEAD_DIM + d] = (__bf16)out;
}

// ---------------------------------------------------------------------------
// Flash attention: per WG one (b, h, 128 q-rows). 8 waves x 16 q-rows each.
// Q frags in registers; K/Vt tiles async-loaded to LDS; per-wave P scratch
// aliases the K-tile region after a barrier.
// ---------------------------------------------------------------------------
__global__ __launch_bounds__(256) void attn_fa(const __bf16* __restrict__ Qb,
                                               const __bf16* __restrict__ Kb,
                                               const __bf16* __restrict__ Vt,
                                               const float* __restrict__ mask,
                                               __bf16* __restrict__ Ob,
                                               float scale) {
    __shared__ __bf16 smK[128 * 128];  // K tile; reused as P scratch (8 x 16x128)
    __shared__ __bf16 smV[128 * 128];  // Vt tile [d=128][k=128]

    const int t = threadIdx.x;
    const int lane = t & 31;
    const int wave = t >> 5;
    const int q0 = blockIdx.x * 128;
    const int h = blockIdx.y;
    const int b = blockIdx.z;

    const size_t headOff = ((size_t)(b * NUM_HEAD + h)) * SEQ * HEAD_DIM;
    const __bf16* Qh = Qb + headOff;
    const __bf16* Kh = Kb + headOff;
    const __bf16* Vh = Vt + headOff;  // [D=128][S=2048]

    // Stage Q tile (128x128) into smK via async, pull this wave's strip.
#pragma unroll
    for (int i = 0; i < 8; ++i) {
        int lin = t + 256 * i;
        int row = lin >> 4, c = (lin & 15) * 8;
        async_load_b128(Qh + (size_t)(q0 + row) * HEAD_DIM + c, lds_off(smK + lin * 8));
    }
    wait_async0();
    __syncthreads();
    v16bf qf[4];
#pragma unroll
    for (int kc = 0; kc < 4; ++kc) qf[kc] = frag_a(smK + wave * 16 * 128, 128, kc * 32);

    float m[8], l[8];
    v8f o[8];
#pragma unroll
    for (int r = 0; r < 8; ++r) { m[r] = -1e30f; l[r] = 0.f; }
#pragma unroll
    for (int n = 0; n < 8; ++n) o[n] = {};

    const int colLane = lane & 15;
    const int rowSel = (lane >> 4) << 3;
    const float LOG2E = 1.44269504088896f;

    for (int kb = 0; kb < SEQ; kb += 128) {
        __syncthreads();  // prior iteration's LDS reads complete (incl. Q frag loads)
#pragma unroll
        for (int i = 0; i < 8; ++i) {  // K tile [kcol][d]
            int lin = t + 256 * i;
            int row = lin >> 4, c = (lin & 15) * 8;
            async_load_b128(Kh + (size_t)(kb + row) * HEAD_DIM + c, lds_off(smK + lin * 8));
        }
#pragma unroll
        for (int i = 0; i < 8; ++i) {  // Vt tile [d][k]
            int lin = t + 256 * i;
            int row = lin >> 4, c = (lin & 15) * 8;
            async_load_b128(Vh + (size_t)row * SEQ + kb + c, lds_off(smV + lin * 8));
        }
        wait_async0();
        __syncthreads();

        // scores: 16 q-rows x 128 k-cols (8 tiles), contract over D=128
        v8f sc[8];
#pragma unroll
        for (int tn = 0; tn < 8; ++tn) {
            v8f a = {};
#pragma unroll
            for (int kc = 0; kc < 4; ++kc) {
                v16bf bfr = frag_b(smK + tn * 16 * 128, 128, kc * 32);
                a = wmma_bf16(qf[kc], bfr, a);
            }
            sc[tn] = a;
        }
        // scale + mask
#pragma unroll
        for (int tn = 0; tn < 8; ++tn)
#pragma unroll
            for (int r = 0; r < 8; ++r) {
                int qr = q0 + wave * 16 + rowSel + r;
                sc[tn][r] = sc[tn][r] * scale +
                            mask[(size_t)qr * SEQ + kb + tn * 16 + colLane];
            }
        // online softmax (rows live in 16-lane halves: shfl_xor 1/2/4/8)
        float alpha[8];
#pragma unroll
        for (int r = 0; r < 8; ++r) {
            float v = sc[0][r];
#pragma unroll
            for (int tn = 1; tn < 8; ++tn) v = fmaxf(v, sc[tn][r]);
            v = fmaxf(v, __shfl_xor(v, 1));
            v = fmaxf(v, __shfl_xor(v, 2));
            v = fmaxf(v, __shfl_xor(v, 4));
            v = fmaxf(v, __shfl_xor(v, 8));
            float nm = fmaxf(m[r], v);
            alpha[r] = exp2f((m[r] - nm) * LOG2E);
            m[r] = nm;
        }
        float rs[8];
#pragma unroll
        for (int r = 0; r < 8; ++r) rs[r] = 0.f;
#pragma unroll
        for (int tn = 0; tn < 8; ++tn)
#pragma unroll
            for (int r = 0; r < 8; ++r) {
                float p = exp2f((sc[tn][r] - m[r]) * LOG2E);
                sc[tn][r] = p;
                rs[r] += p;
            }
#pragma unroll
        for (int r = 0; r < 8; ++r) {
            float s = rs[r];
            s += __shfl_xor(s, 1);
            s += __shfl_xor(s, 2);
            s += __shfl_xor(s, 4);
            s += __shfl_xor(s, 8);
            l[r] = l[r] * alpha[r] + s;
        }
#pragma unroll
        for (int n = 0; n < 8; ++n)
#pragma unroll
            for (int r = 0; r < 8; ++r) o[n][r] *= alpha[r];

        __syncthreads();  // everyone done reading K tile; reuse as P scratch
        __bf16* P = smK + wave * 16 * 128;
#pragma unroll
        for (int tn = 0; tn < 8; ++tn)
#pragma unroll
            for (int r = 0; r < 8; ++r)
                P[(rowSel + r) * 128 + tn * 16 + colLane] = (__bf16)sc[tn][r];

        // O += P @ V  (A = P 16x128, B = Vt rows [d][k])
        v16bf pa[4];
#pragma unroll
        for (int kc = 0; kc < 4; ++kc) pa[kc] = frag_a(P, 128, kc * 32);
#pragma unroll
        for (int n = 0; n < 8; ++n)
#pragma unroll
            for (int kc = 0; kc < 4; ++kc) {
                v16bf bfr = frag_b(smV + n * 16 * 128, 128, kc * 32);
                o[n] = wmma_bf16(pa[kc], bfr, o[n]);
            }
    }

    // normalize + write bf16 [B, S, H*D]
#pragma unroll
    for (int r = 0; r < 8; ++r) l[r] = 1.f / l[r];
#pragma unroll
    for (int n = 0; n < 8; ++n)
#pragma unroll
        for (int r = 0; r < 8; ++r) {
            int srow = q0 + wave * 16 + rowSel + r;
            int d = n * 16 + colLane;
            Ob[((size_t)(b * SEQ) + srow) * DIM + h * HEAD_DIM + d] = (__bf16)(o[n][r] * l[r]);
        }
}

// ---------------------------------------------------------------------------
extern "C" void kernel_launch(void* const* d_in, const int* in_sizes, int n_in,
                              void* d_out, int out_size, void* d_ws, size_t ws_size,
                              hipStream_t stream) {
    const float* x    = (const float*)d_in[0];
    const float* cosT = (const float*)d_in[1];
    const float* sinT = (const float*)d_in[2];
    const float* mask = (const float*)d_in[3];
    const float* wq_w = (const float*)d_in[4];
    const float* wq_b = (const float*)d_in[5];
    const float* wk_w = (const float*)d_in[6];
    const float* wk_b = (const float*)d_in[7];
    const float* wv_w = (const float*)d_in[8];
    const float* wv_b = (const float*)d_in[9];
    const float* wo_w = (const float*)d_in[10];
    const float* wo_b = (const float*)d_in[11];

    char* ws = (char*)d_ws;
    __bf16* xb   = (__bf16*)(ws);                 // 16 MB
    __bf16* wqT  = (__bf16*)(ws + (16ull << 20)); // 8 MB each
    __bf16* wkT  = (__bf16*)(ws + (24ull << 20));
    __bf16* wvT  = (__bf16*)(ws + (32ull << 20));
    __bf16* woT  = (__bf16*)(ws + (40ull << 20));
    float*  Ftmp = (float*)(ws + (48ull << 20));  // 32 MB fp32 scratch
    __bf16* qb   = (__bf16*)(ws + (80ull << 20)); // 16 MB
    __bf16* kb   = (__bf16*)(ws + (96ull << 20));
    __bf16* vtb  = (__bf16*)(ws + (112ull << 20));
    __bf16* ob   = (__bf16*)(ws + (128ull << 20)); // end 144 MB

    // x -> bf16
    convert_f32_bf16<<<(MROWS * DIM) / (256 * 4), 256, 0, stream>>>(x, xb, MROWS * DIM);

    // weights: [K,N] fp32 -> [N,K] bf16
    dim3 wgrid(DIM / 64, DIM / 64, 1);
    transpose_conv<<<wgrid, 256, 0, stream>>>(wq_w, DIM, wqT, DIM, 1, 0, 0, 0);
    transpose_conv<<<wgrid, 256, 0, stream>>>(wk_w, DIM, wkT, DIM, 1, 0, 0, 0);
    transpose_conv<<<wgrid, 256, 0, stream>>>(wv_w, DIM, wvT, DIM, 1, 0, 0, 0);
    transpose_conv<<<wgrid, 256, 0, stream>>>(wo_w, DIM, woT, DIM, 1, 0, 0, 0);

    dim3 ggrid(MROWS / 128, DIM / 256);
    const int ropeGrid = (MROWS * DIM) / 256;

    // Q projection + RoPE
    gemm_bf16<<<ggrid, 256, 0, stream>>>(xb, wqT, wq_b, Ftmp, MROWS, DIM, DIM);
    rope_bf16<<<ropeGrid, 256, 0, stream>>>(Ftmp, cosT, sinT, qb);
    // K projection + RoPE
    gemm_bf16<<<ggrid, 256, 0, stream>>>(xb, wkT, wk_b, Ftmp, MROWS, DIM, DIM);
    rope_bf16<<<ropeGrid, 256, 0, stream>>>(Ftmp, cosT, sinT, kb);
    // V projection + transpose to [B,H,D,S]
    gemm_bf16<<<ggrid, 256, 0, stream>>>(xb, wvT, wv_b, Ftmp, MROWS, DIM, DIM);
    transpose_conv<<<dim3(SEQ / 64, HEAD_DIM / 64, BSZ * NUM_HEAD), 256, 0, stream>>>(
        Ftmp, DIM, vtb, SEQ, NUM_HEAD,
        (size_t)SEQ * DIM, (size_t)HEAD_DIM, (size_t)HEAD_DIM * SEQ);

    // attention
    attn_fa<<<dim3(SEQ / 128, NUM_HEAD, BSZ), 256, 0, stream>>>(
        qb, kb, vtb, mask, ob, 0.08838834764831845f);

    // output projection -> fp32 d_out
    gemm_bf16<<<ggrid, 256, 0, stream>>>(ob, woT, wo_b, (float*)d_out, MROWS, DIM, DIM);
}
